// ProbGAT_6786048328633
// MI455X (gfx1250) — compile-verified, matrix operand order checked by reading, not run on hardware
//
#include <hip/hip_runtime.h>
#include <hip/hip_bf16.h>

#define Hdim 128
#define Ddeg 16

typedef __attribute__((ext_vector_type(16))) __bf16 v16bf;
typedef __attribute__((ext_vector_type(8)))  float  v8f;

// Load one 16x32 bf16 WMMA fragment (A-style) from a row-major [rows][128]
// bf16 tile. Per ISA 7.12.2: lane m (m<16) holds K = {koff..koff+7, 16+koff..16+koff+7}
// with koff = 0 for lanes 0-15 and 8 for lanes 16-31; 2 bf16 packed per VGPR.
// Both halves are 16B-contiguous -> two b128 loads.
__device__ __forceinline__ v16bf load_frag(const __bf16* rowbase, int kt, int koff) {
    union { uint4 q[2]; v16bf v; } f;
    f.q[0] = *reinterpret_cast<const uint4*>(rowbase + kt * 32 + koff);
    f.q[1] = *reinterpret_cast<const uint4*>(rowbase + kt * 32 + 16 + koff);
    return f.v;
}

__device__ __forceinline__ v8f wmma_bf16(v16bf a, v16bf b, v8f c) {
    return __builtin_amdgcn_wmma_f32_16x16x32_bf16(false, a, false, b, (short)0, c, false, false);
}

// Direct global->LDS async copy (16B per lane), GVS addressing:
// SGPR64 base + per-lane VGPR byte offset; LDS dest = low 32 bits of generic ptr.
// Tracked by ASYNCcnt (ISA 10 / 15.18.3, opcode 98).
__device__ __forceinline__ void async_load_b128_to_lds(unsigned lds_off,
                                                       unsigned long long sbase,
                                                       unsigned voff) {
    asm volatile("global_load_async_to_lds_b128 %0, %1, %2"
                 :: "v"(lds_off), "v"(voff), "s"(sbase)
                 : "memory");
}

__device__ __forceinline__ void wait_asynccnt0() {
    asm volatile("s_wait_asynccnt 0x0" ::: "memory");
}

// ---------------- K0: convert + transpose 5 weight matrices to bf16 --------
// out layout: m*16384 + n*128 + k  (i.e. W^T), m = {att_fc1, w0, w1, fc1, fc2}
__global__ void convert_weights_kernel(const float* __restrict__ a1w,
                                       const float* __restrict__ w,
                                       const float* __restrict__ f1w,
                                       const float* __restrict__ f2w,
                                       __bf16* __restrict__ out) {
    int idx = blockIdx.x * blockDim.x + threadIdx.x;
    if (idx >= 5 * 16384) return;
    int m = idx >> 14, r = idx & 16383;
    int n = r >> 7, k = r & 127;
    const float* src = (m == 0) ? a1w : (m == 1) ? w : (m == 2) ? (w + 16384)
                     : (m == 3) ? f1w : f2w;
    out[m * 16384 + n * Hdim + k] = (__bf16)src[k * Hdim + n];
}

// ---------------- K1: per-edge attention logits ----------------------------
// dyn LDS: [16384] att_fc1^T bf16 (32KB) + 8 waves * [16][128] h-tile (4KB each)
__global__ void __launch_bounds__(256) edge_logits_kernel(
    const float* __restrict__ u, const float* __restrict__ x,
    const float* __restrict__ b1, const float* __restrict__ w2,
    const float* __restrict__ b2, const int* __restrict__ ei,
    const __bf16* __restrict__ a1T, float* __restrict__ logits, int E_) {
    extern __shared__ __bf16 smem[];
    __bf16* wlds = smem;  // 16384 bf16
    const int tid = threadIdx.x, lane = tid & 31, wave = tid >> 5;
    __bf16* htile = smem + 16384 + wave * (16 * Hdim);

    {   // async global->LDS copy of att_fc1^T (2048 x 16B), no VGPR round-trip
        unsigned long long src = (unsigned long long)a1T;
        unsigned dstbase = (unsigned)(unsigned long long)(void*)wlds;
        for (int i = tid; i < 2048; i += 256)
            async_load_b128_to_lds(dstbase + (unsigned)i * 16u, src, (unsigned)i * 16u);
        wait_asynccnt0();
    }
    __syncthreads();

    const int e0 = blockIdx.x * 128 + wave * 16;
    const int c0 = lane * 4;
    for (int r = 0; r < 16; ++r) {
        int e = e0 + r;
        float hx = 0.f, hy = 0.f, hz = 0.f, hw = 0.f;
        if (e < E_) {
            int k = ei[e], i = ei[E_ + e];
            float du = u[k] - u[i];
            float4 xk = *(const float4*)(x + (size_t)k * Hdim + c0);
            float4 xi = *(const float4*)(x + (size_t)i * Hdim + c0);
            hx = du * (xk.x - xi.x); hy = du * (xk.y - xi.y);
            hz = du * (xk.z - xi.z); hw = du * (xk.w - xi.w);
        }
        __bf16* t = htile + r * Hdim + c0;
        t[0] = (__bf16)hx; t[1] = (__bf16)hy; t[2] = (__bf16)hz; t[3] = (__bf16)hw;
    }

    const int row  = lane & 15;
    const int koff = (lane >= 16) ? 8 : 0;
    const __bf16* arow = htile + row * Hdim;
    v16bf afrag[4];
#pragma unroll
    for (int kt = 0; kt < 4; ++kt) afrag[kt] = load_frag(arow, kt, koff);

    float partial[8];
#pragma unroll
    for (int v = 0; v < 8; ++v) partial[v] = 0.f;

#pragma unroll
    for (int nt = 0; nt < 8; ++nt) {
        int n = nt * 16 + row;
        const __bf16* brow = wlds + n * Hdim;
        v8f acc = {};
#pragma unroll
        for (int kt = 0; kt < 4; ++kt)
            acc = wmma_bf16(afrag[kt], load_frag(brow, kt, koff), acc);
        float bias = b1[n];
        float wv   = w2[n];
#pragma unroll
        for (int v = 0; v < 8; ++v) {
            float hval = fmaxf(acc[v] + bias, 0.f);
            partial[v] += hval * wv;
        }
    }
    // reduce across the 16 lanes holding the same rows (xor masks < 16)
#pragma unroll
    for (int m = 1; m < 16; m <<= 1)
#pragma unroll
        for (int v = 0; v < 8; ++v) partial[v] += __shfl_xor(partial[v], m, 32);

    if (row == 0) {
        int mbase = (lane >= 16) ? 8 : 0;
        float bb = b2[0];
#pragma unroll
        for (int v = 0; v < 8; ++v) {
            int e = e0 + mbase + v;
            if (e < E_) logits[e] = partial[v] + bb;
        }
    }
}

// ---------------- K2..K5: global softmax reductions ------------------------
__global__ void reduce_max_kernel(const float* __restrict__ l, int n,
                                  float* __restrict__ part) {
    __shared__ float s[256];
    float m = -3.402823466e38f;
    for (int i = blockIdx.x * 256 + threadIdx.x; i < n; i += 256 * 256)
        m = fmaxf(m, l[i]);
    s[threadIdx.x] = m; __syncthreads();
    for (int o = 128; o; o >>= 1) {
        if (threadIdx.x < o) s[threadIdx.x] = fmaxf(s[threadIdx.x], s[threadIdx.x + o]);
        __syncthreads();
    }
    if (threadIdx.x == 0) part[blockIdx.x] = s[0];
}

__global__ void final_max_kernel(const float* __restrict__ part,
                                 float* __restrict__ scal) {
    __shared__ float s[256];
    s[threadIdx.x] = part[threadIdx.x]; __syncthreads();
    for (int o = 128; o; o >>= 1) {
        if (threadIdx.x < o) s[threadIdx.x] = fmaxf(s[threadIdx.x], s[threadIdx.x + o]);
        __syncthreads();
    }
    if (threadIdx.x == 0) scal[0] = s[0];
}

__global__ void reduce_sum_kernel(const float* __restrict__ l, int n,
                                  const float* __restrict__ scal,
                                  float* __restrict__ part) {
    __shared__ float s[256];
    float mx = scal[0];
    float acc = 0.f;
    for (int i = blockIdx.x * 256 + threadIdx.x; i < n; i += 256 * 256)
        acc += __expf(l[i] - mx);
    s[threadIdx.x] = acc; __syncthreads();
    for (int o = 128; o; o >>= 1) {
        if (threadIdx.x < o) s[threadIdx.x] += s[threadIdx.x + o];
        __syncthreads();
    }
    if (threadIdx.x == 0) part[blockIdx.x] = s[0];
}

__global__ void final_sum_kernel(const float* __restrict__ part,
                                 float* __restrict__ scal) {
    __shared__ float s[256];
    s[threadIdx.x] = part[threadIdx.x]; __syncthreads();
    for (int o = 128; o; o >>= 1) {
        if (threadIdx.x < o) s[threadIdx.x] += s[threadIdx.x + o];
        __syncthreads();
    }
    if (threadIdx.x == 0) scal[1] = 1.0f / s[0];
}

// ---------------- K6: aggregation + node-head GEMM chain -------------------
// dyn LDS: 4 weight mats bf16 (128KB) + 8 waves * (xa + agg tiles, 8KB each)
__global__ void __launch_bounds__(256) node_out_kernel(
    const float* __restrict__ x, const int* __restrict__ ei,
    const int* __restrict__ nb, const float* __restrict__ logits,
    const float* __restrict__ scal, const __bf16* __restrict__ wT,
    const float* __restrict__ fc1_b, const float* __restrict__ fc2_b,
    float* __restrict__ out, int N_, int E_) {
    extern __shared__ __bf16 smem[];
    __bf16* w0l = smem;                 // w0^T
    __bf16* w1l = smem + 16384;         // w1^T
    __bf16* f1l = smem + 32768;         // fc1^T
    __bf16* f2l = smem + 49152;         // fc2^T
    const int tid = threadIdx.x, lane = tid & 31, wave = tid >> 5;
    __bf16* xa = smem + 65536 + wave * (2 * 16 * Hdim);
    __bf16* ag = xa + 16 * Hdim;

    {   // async global->LDS copy of 4 weight matrices (8192 x 16B = 128KB)
        unsigned long long src = (unsigned long long)wT;
        unsigned dstbase = (unsigned)(unsigned long long)(void*)smem;
        for (int i = tid; i < 8192; i += 256)
            async_load_b128_to_lds(dstbase + (unsigned)i * 16u, src, (unsigned)i * 16u);
        wait_asynccnt0();
    }
    __syncthreads();

    const float mx = scal[0], inv = scal[1];
    const int n0 = blockIdx.x * 128 + wave * 16;
    const int c0 = lane * 4;

    for (int r = 0; r < 16; ++r) {
        int node = n0 + r;
        float xx = 0.f, xy = 0.f, xz = 0.f, xw = 0.f;
        float ax = 0.f, ay = 0.f, az = 0.f, aw = 0.f;
        if (node < N_) {
            float4 xv = *(const float4*)(x + (size_t)node * Hdim + c0);
            xx = xv.x; xy = xv.y; xz = xv.z; xw = xv.w;
            for (int d = 0; d < Ddeg; ++d) {
                int e = nb[node * Ddeg + d];
                if (e < E_) {
                    int k = ei[e];
                    float a = __expf(logits[e] - mx) * inv;
                    float4 xk = *(const float4*)(x + (size_t)k * Hdim + c0);
                    ax += a * xk.x; ay += a * xk.y; az += a * xk.z; aw += a * xk.w;
                }
            }
        }
        __bf16* tx = xa + r * Hdim + c0;
        tx[0] = (__bf16)xx; tx[1] = (__bf16)xy; tx[2] = (__bf16)xz; tx[3] = (__bf16)xw;
        __bf16* ta = ag + r * Hdim + c0;
        ta[0] = (__bf16)ax; ta[1] = (__bf16)ay; ta[2] = (__bf16)az; ta[3] = (__bf16)aw;
    }

    const int row  = lane & 15;
    const int koff = (lane >= 16) ? 8 : 0;
    const int moff = (lane >= 16) ? 8 : 0;  // C-layout row offset per lane half

    // GEMM1: C = x_tile @ w0 + agg_tile @ w1   (fused into one accumulator)
    v16bf af[4], bf_[4];
#pragma unroll
    for (int kt = 0; kt < 4; ++kt) {
        af[kt]  = load_frag(xa + row * Hdim, kt, koff);
        bf_[kt] = load_frag(ag + row * Hdim, kt, koff);
    }
    v8f C[8];
#pragma unroll
    for (int nt = 0; nt < 8; ++nt) {
        const __bf16* b0row = w0l + (nt * 16 + row) * Hdim;
        const __bf16* b1row = w1l + (nt * 16 + row) * Hdim;
        v8f acc = {};
#pragma unroll
        for (int kt = 0; kt < 4; ++kt)
            acc = wmma_bf16(af[kt], load_frag(b0row, kt, koff), acc);
#pragma unroll
        for (int kt = 0; kt < 4; ++kt)
            acc = wmma_bf16(bf_[kt], load_frag(b1row, kt, koff), acc);
        C[nt] = acc;
    }
    // stage x_out as bf16 (reuse xa; same-wave LDS ordering makes this safe)
#pragma unroll
    for (int nt = 0; nt < 8; ++nt)
#pragma unroll
        for (int v = 0; v < 8; ++v)
            xa[(v + moff) * Hdim + nt * 16 + row] = (__bf16)C[nt][v];

    // GEMM2: y = relu(x_out @ fc1 + fc1_b)
#pragma unroll
    for (int kt = 0; kt < 4; ++kt) af[kt] = load_frag(xa + row * Hdim, kt, koff);
#pragma unroll
    for (int nt = 0; nt < 8; ++nt) {
        const __bf16* brow = f1l + (nt * 16 + row) * Hdim;
        v8f acc = {};
#pragma unroll
        for (int kt = 0; kt < 4; ++kt)
            acc = wmma_bf16(af[kt], load_frag(brow, kt, koff), acc);
        float bias = fc1_b[nt * 16 + row];
#pragma unroll
        for (int v = 0; v < 8; ++v) C[nt][v] = fmaxf(acc[v] + bias, 0.f);
    }
#pragma unroll
    for (int nt = 0; nt < 8; ++nt)
#pragma unroll
        for (int v = 0; v < 8; ++v)
            xa[(v + moff) * Hdim + nt * 16 + row] = (__bf16)C[nt][v];

    // GEMM3: out = y @ fc2 + fc2_b
#pragma unroll
    for (int kt = 0; kt < 4; ++kt) af[kt] = load_frag(xa + row * Hdim, kt, koff);
#pragma unroll
    for (int nt = 0; nt < 8; ++nt) {
        const __bf16* brow = f2l + (nt * 16 + row) * Hdim;
        v8f acc = {};
#pragma unroll
        for (int kt = 0; kt < 4; ++kt)
            acc = wmma_bf16(af[kt], load_frag(brow, kt, koff), acc);
        float bias = fc2_b[nt * 16 + row];
#pragma unroll
        for (int v = 0; v < 8; ++v) {
            int node = n0 + v + moff;
            if (node < N_) out[(size_t)node * Hdim + nt * 16 + row] = acc[v] + bias;
        }
    }
}

// ---------------------------------------------------------------------------
extern "C" void kernel_launch(void* const* d_in, const int* in_sizes, int n_in,
                              void* d_out, int out_size, void* d_ws, size_t ws_size,
                              hipStream_t stream) {
    const float* u   = (const float*)d_in[0];
    const float* x   = (const float*)d_in[1];
    const float* a1w = (const float*)d_in[2];
    const float* a1b = (const float*)d_in[3];
    const float* a2w = (const float*)d_in[4];
    const float* a2b = (const float*)d_in[5];
    const float* w   = (const float*)d_in[6];
    const float* f1w = (const float*)d_in[7];
    const float* f1b = (const float*)d_in[8];
    const float* f2w = (const float*)d_in[9];
    const float* f2b = (const float*)d_in[10];
    const int*   ei  = (const int*)d_in[11];
    const int*   nb  = (const int*)d_in[12];

    const int N_ = in_sizes[0];          // u is [N,1]
    const int E_ = in_sizes[11] / 2;     // edge_index is [2,E]

    // workspace layout
    __bf16* a1T    = (__bf16*)d_ws;               // 16384 bf16
    __bf16* wT     = a1T + 16384;                 // 4 * 16384 bf16 (w0,w1,fc1,fc2)
    float*  logits = (float*)(wT + 4 * 16384);    // E floats
    float*  pmax   = logits + E_;                 // 256
    float*  psum   = pmax + 256;                  // 256
    float*  scal   = psum + 256;                  // [0]=max, [1]=1/sum

    convert_weights_kernel<<<(5 * 16384 + 255) / 256, 256, 0, stream>>>(
        a1w, w, f1w, f2w, a1T);

    edge_logits_kernel<<<(E_ + 127) / 128, 256, 64 * 1024, stream>>>(
        u, x, a1b, a2w, a2b, ei, a1T, logits, E_);

    reduce_max_kernel<<<256, 256, 0, stream>>>(logits, E_, pmax);
    final_max_kernel<<<1, 256, 0, stream>>>(pmax, scal);
    reduce_sum_kernel<<<256, 256, 0, stream>>>(logits, E_, scal, psum);
    final_sum_kernel<<<1, 256, 0, stream>>>(psum, scal);

    node_out_kernel<<<(N_ + 127) / 128, 256, 192 * 1024, stream>>>(
        x, ei, nb, logits, scal, wT, f1b, f2b, (float*)d_out, N_, E_);
}
